// TransformerBlock_60507499266803
// MI455X (gfx1250) — compile-verified, matrix operand order checked by reading, not run on hardware
//
#include <hip/hip_runtime.h>
#include <hip/hip_bf16.h>
#include <stdint.h>

#define Bdim 2
#define Ldim 1024
#define Ddim 1024
#define Hdim 16
#define DHdim 64
#define Edim 8
#define Rdim 8
#define DFFdim 2752
#define Tdim (Bdim * Ldim)
#define LORA_SCALE (1.0f / 8.0f)

typedef __bf16 bf16;
typedef __attribute__((ext_vector_type(16))) __bf16 v16bf;
typedef __attribute__((ext_vector_type(8)))  __bf16 v8bf;
typedef __attribute__((ext_vector_type(8)))  float  v8f;

// ---------------------------------------------------------------------------
// WMMA helpers
// ---------------------------------------------------------------------------
__device__ __forceinline__ v8f wmma_bf16(v16bf a, v16bf b, v8f c) {
  return __builtin_amdgcn_wmma_f32_16x16x32_bf16(false, a, false, b, (short)0, c,
                                                 false, false);
}

// Load a 16x32 bf16 fragment (A-layout / symmetric B-layout): lane holds
// row = lane&15, K = (lane>>4)*8 + [0..7] and +16 + [0..7]  (two b128 loads).
__device__ __forceinline__ v16bf load_frag(const bf16* base, int ld, int lane) {
  const int row = lane & 15;
  const int kb  = (lane >> 4) * 8;
  const bf16* p = base + (size_t)row * ld + kb;
  v8bf lo = *(const v8bf*)(p);
  v8bf hi = *(const v8bf*)(p + 16);
  v16bf r;
#pragma unroll
  for (int i = 0; i < 8; ++i) { r[i] = lo[i]; r[i + 8] = hi[i]; }
  return r;
}

// reductions across the 16-lane half (rows of a 16x16 C tile live per half)
__device__ __forceinline__ float halfmax(float v) {
#pragma unroll
  for (int m = 1; m <= 8; m <<= 1) v = fmaxf(v, __shfl_xor(v, m, 32));
  return v;
}
__device__ __forceinline__ float halfsum(float v) {
#pragma unroll
  for (int m = 1; m <= 8; m <<= 1) v += __shfl_xor(v, m, 32);
  return v;
}

// ---------------------------------------------------------------------------
// 0) hyper-connection mixing weights (tiny)
// mix layout: [0..15]=H1, [16..19]=pre1, [20..23]=post1,
//             [24..39]=H2, [40..43]=pre2, [44..47]=post2
// ---------------------------------------------------------------------------
__device__ __forceinline__ float softmax2_first(float a, float b) {
  float m  = fmaxf(a, b);
  float e0 = __expf(a - m), e1 = __expf(b - m);
  return e0 / (e0 + e1);
}

__global__ void mix_kernel(const float* f1a, const float* f2a, const float* pre1,
                           const float* post1, const float* f1b, const float* f2b,
                           const float* pre2, const float* post2, float* mix) {
  if (threadIdx.x != 0 || blockIdx.x != 0) return;
  float a1 = softmax2_first(f1a[0], f1a[1]);
  float a2 = softmax2_first(f2a[0], f2a[1]);
  float b1 = softmax2_first(f1b[0], f1b[1]);
  float b2 = softmax2_first(f2b[0], f2b[1]);
  float U1[2][2] = {{a1, 1.f - a1}, {1.f - a1, a1}};
  float U2[2][2] = {{a2, 1.f - a2}, {1.f - a2, a2}};
  float V1[2][2] = {{b1, 1.f - b1}, {1.f - b1, b1}};
  float V2[2][2] = {{b2, 1.f - b2}, {1.f - b2, b2}};
  for (int i = 0; i < 2; ++i)
    for (int j = 0; j < 2; ++j)
      for (int k = 0; k < 2; ++k)
        for (int l = 0; l < 2; ++l) {
          mix[(i * 2 + k) * 4 + (j * 2 + l)]      = U1[i][j] * U2[k][l];
          mix[24 + (i * 2 + k) * 4 + (j * 2 + l)] = V1[i][j] * V2[k][l];
        }
  const float* srcs[4] = {pre1, post1, pre2, post2};
  const int offs[4]    = {16, 20, 40, 44};
  for (int s = 0; s < 4; ++s) {
    const float* f = srcs[s];
    float m = fmaxf(fmaxf(f[0], f[1]), fmaxf(f[2], f[3]));
    float e[4], sum = 0.f;
    for (int i = 0; i < 4; ++i) { e[i] = __expf(f[i] - m); sum += e[i]; }
    for (int i = 0; i < 4; ++i) mix[offs[s] + i] = e[i] / sum;
  }
}

// ---------------------------------------------------------------------------
// 1) transpose + cast fp32 (K x N) -> bf16 (N x K)
// ---------------------------------------------------------------------------
__global__ void transpose_cast_kernel(const float* __restrict__ W,
                                      bf16* __restrict__ Wt, int K, int N) {
  int idx = blockIdx.x * 256 + threadIdx.x;
  if (idx >= K * N) return;
  int n = idx / K, k = idx - n * K;
  Wt[(size_t)n * K + k] = (bf16)W[(size_t)k * N + n];
}

// ---------------------------------------------------------------------------
// 2) branch_in = sum_n pre_w[n]*streams[.,n,.] ; RMSNorm -> bf16 (+opt fp32)
//    one block per token, 256 threads, D = 4*256
// ---------------------------------------------------------------------------
__global__ void __launch_bounds__(256)
branch_rms_kernel(const float* __restrict__ src4, const float* __restrict__ prew,
                  const float* __restrict__ normw, bf16* __restrict__ out_bf,
                  float* __restrict__ out_f) {
  __shared__ float red[256];
  const int t = blockIdx.x, tid = threadIdx.x;
  const float* sp = src4 + (size_t)t * 4 * Ddim;
  const float p0 = prew[0], p1 = prew[1], p2 = prew[2], p3 = prew[3];
  float vals[4], ss = 0.f;
#pragma unroll
  for (int i = 0; i < 4; ++i) {
    int d = tid + i * 256;
    float bi = p0 * sp[d] + p1 * sp[Ddim + d] + p2 * sp[2 * Ddim + d] +
               p3 * sp[3 * Ddim + d];
    vals[i] = bi;
    ss += bi * bi;
  }
  red[tid] = ss;
  __syncthreads();
  for (int o = 128; o > 0; o >>= 1) {
    if (tid < o) red[tid] += red[tid + o];
    __syncthreads();
  }
  float nrm = rsqrtf(red[0] * (1.0f / Ddim) + 1e-6f);
#pragma unroll
  for (int i = 0; i < 4; ++i) {
    int d = tid + i * 256;
    float y = vals[i] * nrm * normw[d];
    out_bf[(size_t)t * Ddim + d] = (bf16)y;
    if (out_f) out_f[(size_t)t * Ddim + d] = y;
  }
}

// ---------------------------------------------------------------------------
// 3) bf16 WMMA GEMM: C(MxN) = A(MxK) * Bt(NxK)^T ; wave tile 32x64,
//    4 waves/block stacked in M -> 128x64 block tile.
// ---------------------------------------------------------------------------
__global__ void __launch_bounds__(128)
gemm_bf16_kernel(const bf16* __restrict__ A, const bf16* __restrict__ Bt,
                 float* __restrict__ Cf, bf16* __restrict__ Cb, int M, int N,
                 int K) {
  const int wave = threadIdx.x >> 5;
  const int lane = threadIdx.x & 31;
  const int m0 = blockIdx.x * 128 + wave * 32;
  const int n0 = blockIdx.y * 64;
  v8f acc[2][4];
#pragma unroll
  for (int mt = 0; mt < 2; ++mt)
#pragma unroll
    for (int nt = 0; nt < 4; ++nt)
#pragma unroll
      for (int i = 0; i < 8; ++i) acc[mt][nt][i] = 0.0f;

  for (int k0 = 0; k0 < K; k0 += 32) {
    v16bf a0 = load_frag(A + (size_t)m0 * K + k0, K, lane);
    v16bf a1 = load_frag(A + (size_t)(m0 + 16) * K + k0, K, lane);
#pragma unroll
    for (int nt = 0; nt < 4; ++nt) {
      v16bf b = load_frag(Bt + (size_t)(n0 + nt * 16) * K + k0, K, lane);
      acc[0][nt] = wmma_bf16(a0, b, acc[0][nt]);
      acc[1][nt] = wmma_bf16(a1, b, acc[1][nt]);
    }
  }
  const int col = lane & 15;
  const int rbase = (lane >> 4) * 8;
#pragma unroll
  for (int mt = 0; mt < 2; ++mt)
#pragma unroll
    for (int nt = 0; nt < 4; ++nt)
#pragma unroll
      for (int i = 0; i < 8; ++i) {
        int r = m0 + mt * 16 + rbase + i;
        int c = n0 + nt * 16 + col;
        float v = acc[mt][nt][i];
        if (Cf) Cf[(size_t)r * N + c] = v;
        if (Cb) Cb[(size_t)r * N + c] = (bf16)v;
      }
}

// ---------------------------------------------------------------------------
// 4) RoPE on q,k (from bf16 qkv buffer) + V transpose to (b,h,dh,l)
// ---------------------------------------------------------------------------
__global__ void rope_kernel(const bf16* __restrict__ qkv,
                            const float* __restrict__ cosb,
                            const float* __restrict__ sinb, bf16* __restrict__ Qo,
                            bf16* __restrict__ Ko, bf16* __restrict__ Vt) {
  int idx = blockIdx.x * 256 + threadIdx.x;  // (b,h,l,dh2) ; dh2 in [0,32)
  int dh = idx & 31;
  int l = (idx >> 5) & (Ldim - 1);
  int h = (idx >> 15) & (Hdim - 1);
  int b = idx >> 19;
  int t = b * Ldim + l;
  const bf16* base = qkv + (size_t)t * (3 * Ddim);
  int hoff = h * DHdim;
  float c = cosb[l * 32 + dh], s = sinb[l * 32 + dh];
  float q1 = (float)base[hoff + dh], q2 = (float)base[hoff + dh + 32];
  float k1 = (float)base[Ddim + hoff + dh], k2 = (float)base[Ddim + hoff + dh + 32];
  size_t qb = ((size_t)(b * Hdim + h) * Ldim + l) * DHdim;
  Qo[qb + dh]      = (bf16)(q1 * c - q2 * s);
  Qo[qb + dh + 32] = (bf16)(q2 * c + q1 * s);
  Ko[qb + dh]      = (bf16)(k1 * c - k2 * s);
  Ko[qb + dh + 32] = (bf16)(k2 * c + k1 * s);
  float v1 = (float)base[2 * Ddim + hoff + dh];
  float v2 = (float)base[2 * Ddim + hoff + dh + 32];
  size_t vb = (size_t)(b * Hdim + h) * DHdim * Ldim;
  Vt[vb + (size_t)dh * Ldim + l]        = (bf16)v1;
  Vt[vb + (size_t)(dh + 32) * Ldim + l] = (bf16)v2;
}

// ---------------------------------------------------------------------------
// 5) causal flash attention; one wave per 32-row q tile, 8 waves/block.
//    Q,K: (b,h,l,dh) bf16 ; Vt: (b,h,dh,l) bf16 ; Out: (t, h*DH+dh) bf16
// ---------------------------------------------------------------------------
__global__ void __launch_bounds__(256)
attn_kernel(const bf16* __restrict__ Q, const bf16* __restrict__ Kmat,
            const bf16* __restrict__ Vt, bf16* __restrict__ Out) {
  __shared__ alignas(16) bf16 Pst[8][32 * 32];
  const int wave = threadIdx.x >> 5;
  const int lane = threadIdx.x & 31;
  const int tile = blockIdx.x * 8 + wave;
  const int qt = tile & (Ldim / 32 - 1);
  const int bh = tile >> 5;  // L/32 == 32 q-tiles per (b,h)
  const int h = bh & (Hdim - 1);
  const int b = bh >> 4;
  const bf16* Qb = Q + (size_t)bh * Ldim * DHdim;
  const bf16* Kb = Kmat + (size_t)bh * Ldim * DHdim;
  const bf16* Vb = Vt + (size_t)bh * DHdim * Ldim;
  const int q0 = qt * 32;
  const int colh = lane & 15, rbase = (lane >> 4) * 8;

  v16bf qf[2][2];
#pragma unroll
  for (int mt = 0; mt < 2; ++mt)
#pragma unroll
    for (int ks = 0; ks < 2; ++ks)
      qf[mt][ks] = load_frag(Qb + (size_t)(q0 + mt * 16) * DHdim + ks * 32, DHdim, lane);

  v8f o[2][4];
  float mrow[2][8], srow[2][8];
#pragma unroll
  for (int mt = 0; mt < 2; ++mt) {
#pragma unroll
    for (int nt = 0; nt < 4; ++nt)
#pragma unroll
      for (int i = 0; i < 8; ++i) o[mt][nt][i] = 0.f;
#pragma unroll
    for (int i = 0; i < 8; ++i) { mrow[mt][i] = -3.0e38f; srow[mt][i] = 0.f; }
  }
  const float scl = 0.125f;  // 1/sqrt(64)
  bf16* P = Pst[wave];

  for (int kt = 0; kt <= qt; ++kt) {
    const int k0 = kt * 32;
    v8f s[2][2];
#pragma unroll
    for (int mt = 0; mt < 2; ++mt)
#pragma unroll
      for (int ct = 0; ct < 2; ++ct)
#pragma unroll
        for (int i = 0; i < 8; ++i) s[mt][ct][i] = 0.f;

    v16bf kf0a = load_frag(Kb + (size_t)k0 * DHdim + 0, DHdim, lane);
    v16bf kf0b = load_frag(Kb + (size_t)k0 * DHdim + 32, DHdim, lane);
    v16bf kf1a = load_frag(Kb + (size_t)(k0 + 16) * DHdim + 0, DHdim, lane);
    v16bf kf1b = load_frag(Kb + (size_t)(k0 + 16) * DHdim + 32, DHdim, lane);
#pragma unroll
    for (int mt = 0; mt < 2; ++mt) {
      s[mt][0] = wmma_bf16(qf[mt][0], kf0a, s[mt][0]);
      s[mt][0] = wmma_bf16(qf[mt][1], kf0b, s[mt][0]);
      s[mt][1] = wmma_bf16(qf[mt][0], kf1a, s[mt][1]);
      s[mt][1] = wmma_bf16(qf[mt][1], kf1b, s[mt][1]);
    }
#pragma unroll
    for (int mt = 0; mt < 2; ++mt) {
#pragma unroll
      for (int i = 0; i < 8; ++i) {
        int r = q0 + mt * 16 + rbase + i;
        float v0 = s[mt][0][i] * scl;
        float v1 = s[mt][1][i] * scl;
        if (k0 + colh > r) v0 = -1.0e30f;
        if (k0 + 16 + colh > r) v1 = -1.0e30f;
        float mx = halfmax(fmaxf(v0, v1));
        float mn = fmaxf(mrow[mt][i], mx);
        float alpha = __expf(mrow[mt][i] - mn);
        float p0 = __expf(v0 - mn);
        float p1 = __expf(v1 - mn);
        float rs = halfsum(p0 + p1);
        srow[mt][i] = srow[mt][i] * alpha + rs;
        mrow[mt][i] = mn;
#pragma unroll
        for (int nt = 0; nt < 4; ++nt) o[mt][nt][i] *= alpha;
        int lr = mt * 16 + rbase + i;
        P[lr * 32 + colh]      = (bf16)p0;
        P[lr * 32 + 16 + colh] = (bf16)p1;
      }
    }
    asm volatile("s_wait_dscnt 0" ::: "memory");  // wave-local LDS RAW fence
    v16bf pf0 = load_frag(P, 32, lane);
    v16bf pf1 = load_frag(P + 16 * 32, 32, lane);
#pragma unroll
    for (int nt = 0; nt < 4; ++nt) {
      v16bf vf = load_frag(Vb + (size_t)(nt * 16) * Ldim + k0, Ldim, lane);
      o[0][nt] = wmma_bf16(pf0, vf, o[0][nt]);
      o[1][nt] = wmma_bf16(pf1, vf, o[1][nt]);
    }
  }
#pragma unroll
  for (int mt = 0; mt < 2; ++mt)
#pragma unroll
    for (int nt = 0; nt < 4; ++nt)
#pragma unroll
      for (int i = 0; i < 8; ++i) {
        int l = q0 + mt * 16 + rbase + i;
        int c = h * DHdim + nt * 16 + colh;
        float v = o[mt][nt][i] / srow[mt][i];
        Out[(size_t)(b * Ldim + l) * Ddim + c] = (bf16)v;
      }
}

// ---------------------------------------------------------------------------
// 6) hyper combine: dst[t,i,d] = sum_j H[i][j]*src4[t,j,d] + branch[t,d]*post[i]
// ---------------------------------------------------------------------------
__global__ void hyper_combine_kernel(const float* __restrict__ src4,
                                     const float* __restrict__ branch,
                                     const float* __restrict__ Hm,
                                     const float* __restrict__ postw,
                                     float* __restrict__ dst4) {
  int idx = blockIdx.x * 256 + threadIdx.x;  // t*D + d
  int t = idx >> 10, d = idx & (Ddim - 1);
  const float* sp = src4 + (size_t)t * 4 * Ddim + d;
  float sv[4];
#pragma unroll
  for (int j = 0; j < 4; ++j) sv[j] = sp[(size_t)j * Ddim];
  float bo = branch[idx];
  float* op = dst4 + (size_t)t * 4 * Ddim + d;
#pragma unroll
  for (int i = 0; i < 4; ++i) {
    float a = bo * postw[i];
#pragma unroll
    for (int j = 0; j < 4; ++j) a += Hm[i * 4 + j] * sv[j];
    op[(size_t)i * Ddim] = a;
  }
}

// ---------------------------------------------------------------------------
// 7) router: sigmoid scores, top-2 on biased scores, normalized weights
// ---------------------------------------------------------------------------
__global__ void router_kernel(const float* __restrict__ xn,
                              const float* __restrict__ Wr,
                              const float* __restrict__ ebias,
                              float* __restrict__ expw) {
  int t = blockIdx.x * 64 + threadIdx.x;
  if (t >= Tdim) return;
  float acc[Edim];
#pragma unroll
  for (int e = 0; e < Edim; ++e) acc[e] = 0.f;
  const float* x = xn + (size_t)t * Ddim;
  for (int d = 0; d < Ddim; ++d) {
    float xv = x[d];
#pragma unroll
    for (int e = 0; e < Edim; ++e) acc[e] += xv * Wr[d * Edim + e];
  }
  float sc[Edim], bi[Edim];
#pragma unroll
  for (int e = 0; e < Edim; ++e) {
    sc[e] = 1.f / (1.f + __expf(-acc[e]));
    bi[e] = sc[e] + ebias[e];
  }
  int i1 = 0;
#pragma unroll
  for (int e = 1; e < Edim; ++e)
    if (bi[e] > bi[i1]) i1 = e;
  int i2 = (i1 == 0) ? 1 : 0;
#pragma unroll
  for (int e = 0; e < Edim; ++e)
    if (e != i1 && bi[e] > bi[i2]) i2 = e;
  float s1 = sc[i1], s2 = sc[i2];
  float inv = 1.f / (s1 + s2 + 1e-8f);
#pragma unroll
  for (int e = 0; e < Edim; ++e) expw[(size_t)t * Edim + e] = 0.f;
  expw[(size_t)t * Edim + i1] = s1 * inv;
  expw[(size_t)t * Edim + i2] = s2 * inv;
}

// ---------------------------------------------------------------------------
// 8) LoRA down-projection coefficients: U[t, 0..7]  = SCALE * x . sA[:,r]
//                                       U[t, 8+e*8+r] = SCALE*w[t,e] * x . eA[e,:,r]
// ---------------------------------------------------------------------------
__global__ void lora_u_kernel(const float* __restrict__ xn, int Kdim,
                              const float* __restrict__ sA,
                              const float* __restrict__ eA,
                              const float* __restrict__ expw,
                              float* __restrict__ U) {
  int idx = blockIdx.x * 256 + threadIdx.x;  // T * 72
  if (idx >= Tdim * 72) return;
  int t = idx / 72, k = idx - t * 72;
  const float* x = xn + (size_t)t * Kdim;
  float acc = 0.f;
  if (k < Rdim) {
    for (int d = 0; d < Kdim; ++d) acc += x[d] * sA[(size_t)d * Rdim + k];
    acc *= LORA_SCALE;
  } else {
    int e = (k - Rdim) / Rdim, r = (k - Rdim) % Rdim;
    const float* Ae = eA + (size_t)e * Kdim * Rdim;
    for (int d = 0; d < Kdim; ++d) acc += x[d] * Ae[(size_t)d * Rdim + r];
    acc *= LORA_SCALE * expw[(size_t)t * Edim + e];
  }
  U[idx] = acc;
}

// ---------------------------------------------------------------------------
// 9) LoRA up-projection add: Y[t,j] += sum_r U[t,r]*sB[r,j] + sum_k U[t,8+k]*eB[k,j]
// ---------------------------------------------------------------------------
__global__ void lora_add_kernel(float* __restrict__ Y, int N,
                                const float* __restrict__ sB,
                                const float* __restrict__ eB,
                                const float* __restrict__ U) {
  int idx = blockIdx.x * 256 + threadIdx.x;
  if (idx >= Tdim * N) return;
  int t = idx / N, j = idx - t * N;
  const float* u = U + (size_t)t * 72;
  float acc = 0.f;
#pragma unroll
  for (int r = 0; r < Rdim; ++r) acc += u[r] * sB[(size_t)r * N + j];
  for (int k = 0; k < Edim * Rdim; ++k) acc += u[8 + k] * eB[(size_t)k * N + j];
  Y[idx] += acc;
}

// ---------------------------------------------------------------------------
// 10) hidden = silu(gate) * up  (fp32 + bf16); hf may alias gate
// ---------------------------------------------------------------------------
__global__ void hidden_kernel(const float* __restrict__ gate,
                              const float* __restrict__ up,
                              float* __restrict__ hf, bf16* __restrict__ hb) {
  int idx = blockIdx.x * 256 + threadIdx.x;
  if (idx >= Tdim * DFFdim) return;
  float g = gate[idx];
  float h = (g / (1.f + __expf(-g))) * up[idx];
  hf[idx] = h;
  hb[idx] = (bf16)h;
}

// ---------------------------------------------------------------------------
// host orchestration
// ---------------------------------------------------------------------------
extern "C" void kernel_launch(void* const* d_in, const int* in_sizes, int n_in,
                              void* d_out, int out_size, void* d_ws, size_t ws_size,
                              hipStream_t stream) {
  (void)in_sizes; (void)n_in; (void)out_size; (void)ws_size;
  const float* streams     = (const float*)d_in[0];
  const float* norm1_w     = (const float*)d_in[1];
  const float* norm2_w     = (const float*)d_in[2];
  const float* Wqkv        = (const float*)d_in[3];
  const float* Wo          = (const float*)d_in[4];
  const float* rope_cos    = (const float*)d_in[5];
  const float* rope_sin    = (const float*)d_in[6];
  const float* hc1_f1      = (const float*)d_in[7];
  const float* hc1_f2      = (const float*)d_in[8];
  const float* hc1_pre     = (const float*)d_in[9];
  const float* hc1_post    = (const float*)d_in[10];
  const float* hc2_f1      = (const float*)d_in[11];
  const float* hc2_f2      = (const float*)d_in[12];
  const float* hc2_pre     = (const float*)d_in[13];
  const float* hc2_post    = (const float*)d_in[14];
  const float* W_gate      = (const float*)d_in[15];
  const float* W_up        = (const float*)d_in[16];
  const float* W_down      = (const float*)d_in[17];
  const float* sA_gate     = (const float*)d_in[18];
  const float* sB_gate     = (const float*)d_in[19];
  const float* sA_up       = (const float*)d_in[20];
  const float* sB_up       = (const float*)d_in[21];
  const float* sA_down     = (const float*)d_in[22];
  const float* sB_down     = (const float*)d_in[23];
  const float* eA_gate     = (const float*)d_in[24];
  const float* eB_gate     = (const float*)d_in[25];
  const float* eA_up       = (const float*)d_in[26];
  const float* eB_up       = (const float*)d_in[27];
  const float* eA_down     = (const float*)d_in[28];
  const float* eB_down     = (const float*)d_in[29];
  const float* W_router    = (const float*)d_in[30];
  const float* expert_bias = (const float*)d_in[31];
  float* out = (float*)d_out;

  uint8_t* wsp = (uint8_t*)d_ws;
  auto alloc = [&](size_t bytes) -> void* {
    void* r = (void*)wsp;
    wsp += (bytes + 255) & ~((size_t)255);
    return r;
  };

  float* mix     = (float*)alloc(64 * sizeof(float));
  bf16*  xn1_bf  = (bf16*)alloc((size_t)Tdim * Ddim * sizeof(bf16));
  bf16*  Wqkv_t  = (bf16*)alloc((size_t)3 * Ddim * Ddim * sizeof(bf16));
  bf16*  qkv_bf  = (bf16*)alloc((size_t)Tdim * 3 * Ddim * sizeof(bf16));
  bf16*  q_bf    = (bf16*)alloc((size_t)Tdim * Ddim * sizeof(bf16));
  bf16*  k_bf    = (bf16*)alloc((size_t)Tdim * Ddim * sizeof(bf16));
  bf16*  vt_bf   = (bf16*)alloc((size_t)Tdim * Ddim * sizeof(bf16));
  bf16*  attn_bf = (bf16*)alloc((size_t)Tdim * Ddim * sizeof(bf16));
  bf16*  Wo_t    = (bf16*)alloc((size_t)Ddim * Ddim * sizeof(bf16));
  float* branch1 = (float*)alloc((size_t)Tdim * Ddim * sizeof(float));
  float* x1      = (float*)alloc((size_t)Tdim * 4 * Ddim * sizeof(float));
  float* xn2_f   = (float*)alloc((size_t)Tdim * Ddim * sizeof(float));
  bf16*  xn2_bf  = (bf16*)alloc((size_t)Tdim * Ddim * sizeof(bf16));
  bf16*  Wg_t    = (bf16*)alloc((size_t)DFFdim * Ddim * sizeof(bf16));
  bf16*  Wu_t    = (bf16*)alloc((size_t)DFFdim * Ddim * sizeof(bf16));
  bf16*  Wd_t    = (bf16*)alloc((size_t)Ddim * DFFdim * sizeof(bf16));
  float* expw    = (float*)alloc((size_t)Tdim * Edim * sizeof(float));
  float* ug      = (float*)alloc((size_t)Tdim * 72 * sizeof(float));
  float* uu      = (float*)alloc((size_t)Tdim * 72 * sizeof(float));
  float* gate    = (float*)alloc((size_t)Tdim * DFFdim * sizeof(float));
  float* up      = (float*)alloc((size_t)Tdim * DFFdim * sizeof(float));
  bf16*  hid_bf  = (bf16*)alloc((size_t)Tdim * DFFdim * sizeof(bf16));
  float* ud      = (float*)alloc((size_t)Tdim * 72 * sizeof(float));
  float* ffn     = (float*)alloc((size_t)Tdim * Ddim * sizeof(float));

  // mixing weights
  mix_kernel<<<1, 1, 0, stream>>>(hc1_f1, hc1_f2, hc1_pre, hc1_post, hc2_f1,
                                  hc2_f2, hc2_pre, hc2_post, mix);
  // weight transpose+cast
  {
    int n;
    n = Ddim * 3 * Ddim;
    transpose_cast_kernel<<<(n + 255) / 256, 256, 0, stream>>>(Wqkv, Wqkv_t, Ddim, 3 * Ddim);
    n = Ddim * Ddim;
    transpose_cast_kernel<<<(n + 255) / 256, 256, 0, stream>>>(Wo, Wo_t, Ddim, Ddim);
    n = Ddim * DFFdim;
    transpose_cast_kernel<<<(n + 255) / 256, 256, 0, stream>>>(W_gate, Wg_t, Ddim, DFFdim);
    transpose_cast_kernel<<<(n + 255) / 256, 256, 0, stream>>>(W_up, Wu_t, Ddim, DFFdim);
    transpose_cast_kernel<<<(n + 255) / 256, 256, 0, stream>>>(W_down, Wd_t, DFFdim, Ddim);
  }
  // ---- layer 1: attention branch ----
  branch_rms_kernel<<<Tdim, 256, 0, stream>>>(streams, mix + 16, norm1_w, xn1_bf,
                                              (float*)nullptr);
  gemm_bf16_kernel<<<dim3(Tdim / 128, (3 * Ddim) / 64), 128, 0, stream>>>(
      xn1_bf, Wqkv_t, (float*)nullptr, qkv_bf, Tdim, 3 * Ddim, Ddim);
  rope_kernel<<<(Bdim * Hdim * Ldim * 32) / 256, 256, 0, stream>>>(
      qkv_bf, rope_cos, rope_sin, q_bf, k_bf, vt_bf);
  attn_kernel<<<(Bdim * Hdim * (Ldim / 32)) / 8, 256, 0, stream>>>(q_bf, k_bf,
                                                                   vt_bf, attn_bf);
  gemm_bf16_kernel<<<dim3(Tdim / 128, Ddim / 64), 128, 0, stream>>>(
      attn_bf, Wo_t, branch1, (bf16*)nullptr, Tdim, Ddim, Ddim);
  hyper_combine_kernel<<<(Tdim * Ddim) / 256, 256, 0, stream>>>(streams, branch1,
                                                                mix + 0, mix + 20, x1);
  // ---- layer 2: MoL FFN branch ----
  branch_rms_kernel<<<Tdim, 256, 0, stream>>>(x1, mix + 40, norm2_w, xn2_bf, xn2_f);
  router_kernel<<<Tdim / 64, 64, 0, stream>>>(xn2_f, W_router, expert_bias, expw);
  lora_u_kernel<<<(Tdim * 72 + 255) / 256, 256, 0, stream>>>(xn2_f, Ddim, sA_gate,
                                                             eA_gate, expw, ug);
  lora_u_kernel<<<(Tdim * 72 + 255) / 256, 256, 0, stream>>>(xn2_f, Ddim, sA_up,
                                                             eA_up, expw, uu);
  gemm_bf16_kernel<<<dim3(Tdim / 128, DFFdim / 64), 128, 0, stream>>>(
      xn2_bf, Wg_t, gate, (bf16*)nullptr, Tdim, DFFdim, Ddim);
  gemm_bf16_kernel<<<dim3(Tdim / 128, DFFdim / 64), 128, 0, stream>>>(
      xn2_bf, Wu_t, up, (bf16*)nullptr, Tdim, DFFdim, Ddim);
  lora_add_kernel<<<(Tdim * DFFdim + 255) / 256, 256, 0, stream>>>(gate, DFFdim,
                                                                   sB_gate, eB_gate, ug);
  lora_add_kernel<<<(Tdim * DFFdim + 255) / 256, 256, 0, stream>>>(up, DFFdim,
                                                                   sB_up, eB_up, uu);
  hidden_kernel<<<(Tdim * DFFdim + 255) / 256, 256, 0, stream>>>(gate, up, gate,
                                                                 hid_bf);
  lora_u_kernel<<<(Tdim * 72 + 255) / 256, 256, 0, stream>>>(gate, DFFdim, sA_down,
                                                             eA_down, expw, ud);
  gemm_bf16_kernel<<<dim3(Tdim / 128, Ddim / 64), 128, 0, stream>>>(
      hid_bf, Wd_t, ffn, (bf16*)nullptr, Tdim, Ddim, DFFdim);
  lora_add_kernel<<<(Tdim * Ddim + 255) / 256, 256, 0, stream>>>(ffn, Ddim, sB_down,
                                                                 eB_down, ud);
  hyper_combine_kernel<<<(Tdim * Ddim) / 256, 256, 0, stream>>>(x1, ffn, mix + 24,
                                                                mix + 44, out);
}
